// PGN_60232621359349
// MI455X (gfx1250) — matrix operand order repaired and checked by reading, not compiled, not run on hardware
//
#include <hip/hip_runtime.h>
#include <hip/hip_bf16.h>

// PGN message-passing layer, B=8, N=256, H=128, fp32 I/O, bf16 WMMA compute.
// Edge GEMM + masked max fused, edge tiles staged via async global->LDS
// (ASYNCcnt) with double buffering.

#define BB 8
#define NN 256
#define HH 128
#define NEG_BIG (-1.0e6f)

typedef __bf16 bf16_t;
typedef unsigned long long u64;
typedef __attribute__((ext_vector_type(16))) __bf16 v16bf;
typedef __attribute__((ext_vector_type(8)))  __bf16 v8bf;
typedef __attribute__((ext_vector_type(8)))  float  v8f;
typedef __attribute__((ext_vector_type(4)))  int    v4i;

#define AS3 __attribute__((address_space(3)))

// ---------------------------------------------------------------------------
// CDNA5 async global->LDS copy (16 bytes per lane), ASYNCcnt-tracked.
// ROCm clang-22 signature takes int4* generic pointers (src, dst, off, cpol).
// ---------------------------------------------------------------------------
__device__ __forceinline__ void async_b128(const float* gsrc, float* ldst) {
#if defined(__gfx1250__) && __has_builtin(__builtin_amdgcn_global_load_async_to_lds_b128)
  __builtin_amdgcn_global_load_async_to_lds_b128(
      (v4i*)gsrc, (v4i*)ldst, 0, 0);
#else
  unsigned loff = (unsigned)(size_t)(AS3 char*)ldst;   // LDS byte offset
  asm volatile("global_load_async_to_lds_b128 %0, %1, off"
               :: "v"(loff), "v"((u64)(size_t)gsrc) : "memory");
#endif
}

__device__ __forceinline__ void wait_async0() {
#if defined(__gfx1250__) && __has_builtin(__builtin_amdgcn_s_wait_asynccnt)
  __builtin_amdgcn_s_wait_asynccnt(0);
#else
  asm volatile("s_wait_asynccnt 0x0" ::: "memory");
#endif
}

// ---------------------------------------------------------------------------
// A fragment (16x32 bf16, MxK) from a row-major bf16 LDS tile.
// Lanes 0-15/16-31 hold M=lane&15; slot s holds
// K = k0 + s + (s>=8 ? 8:0) + (lane>=16 ? 8:0) -> two 16B LDS reads per lane.
// ---------------------------------------------------------------------------
__device__ __forceinline__ v16bf load_a_frag(const bf16_t* lds, int k0, int rowStride) {
  int lane = threadIdx.x & 31;
  int m    = lane & 15;
  int hi   = (lane >> 4) & 1;
  const bf16_t* p = lds + m * rowStride + k0 + hi * 8;
  v8bf lo = *(const v8bf*)(p);
  v8bf hg = *(const v8bf*)(p + 16);
  return __builtin_shufflevector(lo, hg, 0,1,2,3,4,5,6,7,8,9,10,11,12,13,14,15);
}

// Same fragment but from an f32 LDS tile (async-staged); convert at load time.
__device__ __forceinline__ v16bf load_a_frag_f32lds(const float* lds, int k0, int rowStride) {
  int lane = threadIdx.x & 31;
  int m    = lane & 15;
  int hi   = (lane >> 4) & 1;
  const float* p = lds + m * rowStride + k0 + hi * 8;
  v8f lo = *(const v8f*)(p);
  v8f hg = *(const v8f*)(p + 16);
  v16bf f;
#pragma unroll
  for (int s = 0; s < 8; ++s) { f[s] = (bf16_t)lo[s]; f[8 + s] = (bf16_t)hg[s]; }
  return f;
}

// B fragment (32x16 bf16, KxN) from row-major f32 weight [Ktot,128].
__device__ __forceinline__ v16bf load_b_frag_f32(const float* __restrict__ W,
                                                 int k0, int col, int ldw) {
  int lane = threadIdx.x & 31;
  int koff = k0 + ((lane >> 4) & 1) * 8;
  v16bf f;
#pragma unroll
  for (int s = 0; s < 16; ++s)
    f[s] = (bf16_t)W[(koff + s + (s >= 8 ? 8 : 0)) * ldw + col];
  return f;
}

__device__ __forceinline__ v8f wmma_bf16(v16bf a, v16bf b, v8f c) {
  return __builtin_amdgcn_wmma_f32_16x16x32_bf16(
      false, a, false, b, (short)0, c, false, false);
}

// ---------------------------------------------------------------------------
// Kernel G: msg_g[b,h] = graph_fts[b,:] @ mg_w + mg_b   (8x128, negligible)
// ---------------------------------------------------------------------------
__global__ void pgn_msg_g_kernel(const float* __restrict__ gfts,
                                 const float* __restrict__ mgw,
                                 const float* __restrict__ mgb,
                                 float* __restrict__ msg_g) {
  int b = blockIdx.x, h = threadIdx.x;
  float acc = mgb[h];
#pragma unroll 8
  for (int k = 0; k < HH; ++k) acc += gfts[b * HH + k] * mgw[k * HH + h];
  msg_g[b * HH + h] = acc;
}

// ---------------------------------------------------------------------------
// Kernel A: three z-GEMMs (M=2048, K=256, N=128), z = [node | hidden].
// blockIdx.y: 0 -> msg_1, 1 -> s2 = msg_2 + msg_g, 2 -> h_1
// ---------------------------------------------------------------------------
__global__ void pgn_small_gemm_kernel(const float* __restrict__ node,
                                      const float* __restrict__ hidden,
                                      const float* __restrict__ w1, const float* __restrict__ b1,
                                      const float* __restrict__ w2, const float* __restrict__ b2,
                                      const float* __restrict__ w3, const float* __restrict__ b3,
                                      const float* __restrict__ msg_g,
                                      float* __restrict__ out1, float* __restrict__ out2,
                                      float* __restrict__ out3) {
  __shared__ bf16_t zt[16 * 256];
  const int row0 = blockIdx.x * 16, which = blockIdx.y;
  const int tid = threadIdx.x;

#pragma unroll
  for (int e = 0; e < 16; ++e) {
    int idx = tid * 16 + e;
    int r = idx >> 8, c = idx & 255;
    float v = (c < HH) ? node[(size_t)(row0 + r) * HH + c]
                       : hidden[(size_t)(row0 + r) * HH + (c - HH)];
    zt[idx] = (bf16_t)v;
  }
  __syncthreads();

  const float* W  = (which == 0) ? w1 : (which == 1) ? w2 : w3;
  const float* Bb = (which == 0) ? b1 : (which == 1) ? b2 : b3;
  float*       O  = (which == 0) ? out1 : (which == 1) ? out2 : out3;

  const int wave = tid >> 5, lane = tid & 31;
  const int col  = wave * 16 + (lane & 15);
  const int hi   = (lane >> 4) & 1;

  v8f acc;
#pragma unroll
  for (int v = 0; v < 8; ++v) acc[v] = 0.0f;
#pragma unroll
  for (int k0 = 0; k0 < 256; k0 += 32) {
    v16bf a  = load_a_frag(zt, k0, 256);
    v16bf bb = load_b_frag_f32(W, k0, col, HH);
    acc = wmma_bf16(a, bb, acc);
  }
  const float bias = Bb[col];
#pragma unroll
  for (int v = 0; v < 8; ++v) {
    int r = row0 + v + hi * 8;
    float val = acc[v] + bias;
    if (which == 1) val += msg_g[(r / NN) * HH + col];
    O[(size_t)r * HH + col] = val;
  }
}

// ---------------------------------------------------------------------------
// Kernel B: fused edge GEMM + masked max over source axis i.
// grid = (j-tile=16, b=8, i-half=2), block = 256 (8 waves).
// Edge tile (16x128 f32 = 8KB, contiguous), s2 row and adj row are staged with
// async global->LDS loads, double buffered; WMMA consumes the previous buffer.
// ---------------------------------------------------------------------------
#define TILE_F   2048                     // 16 rows x 128 k (floats)
#define S2_OFF   2048                     // +128 floats
#define ADJ_OFF  2176                     // +16 floats
#define BUF_F    2192

__global__ void pgn_edge_max_kernel(const float* __restrict__ edge,
                                    const float* __restrict__ adj,
                                    const float* __restrict__ s2,
                                    const float* __restrict__ me_w,
                                    float* __restrict__ partial) {
  __shared__ float buf[2][BUF_F];

  const int jt = blockIdx.x, b = blockIdx.y, half = blockIdx.z;
  const int j0 = jt * 16;
  const int tid = threadIdx.x, lane = tid & 31;
  const int col = (tid >> 5) * 16 + (lane & 15);
  const int hi  = (lane >> 4) & 1;
  const int i0  = half * 128;

  // me_w column fragments are i-invariant: preload all 4 K-steps.
  v16bf bfrag[4];
#pragma unroll
  for (int kk = 0; kk < 4; ++kk) bfrag[kk] = load_b_frag_f32(me_w, kk * 32, col, HH);

  v8f mx;
#pragma unroll
  for (int v = 0; v < 8; ++v) mx[v] = NEG_BIG;

  // issue async loads of tile/s2row/adjrow for source node i into buf[slot]
  auto issue = [&](int slot, int i) {
    const float* src = edge + (((size_t)b * NN + i) * NN + j0) * HH;
    float* d = &buf[slot][0];
    async_b128(src + tid * 8,     d + tid * 8);       // 256 thr x 32B = 8KB
    async_b128(src + tid * 8 + 4, d + tid * 8 + 4);
    if (tid < 32)
      async_b128(s2 + ((size_t)b * NN + i) * HH + tid * 4, d + S2_OFF + tid * 4);
    if (tid < 4)
      async_b128(adj + ((size_t)b * NN + i) * NN + j0 + tid * 4, d + ADJ_OFF + tid * 4);
  };

  issue(0, i0);                                       // prologue

  for (int ii = 0; ii < 128; ++ii) {
    const int slot = ii & 1;
    wait_async0();                  // my async writes for buf[slot] done
    __syncthreads();                // everyone's writes visible; prev reads done
    if (ii + 1 < 128) issue(slot ^ 1, i0 + ii + 1);   // overlap next tile

    const float* cur = &buf[slot][0];
    v8f d;
#pragma unroll
    for (int v = 0; v < 8; ++v) d[v] = 0.0f;
#pragma unroll
    for (int kk = 0; kk < 4; ++kk) {
      v16bf a = load_a_frag_f32lds(cur, kk * 32, HH);
      d = wmma_bf16(a, bfrag[kk], d);
    }
    const float s2v = cur[S2_OFF + col];
#pragma unroll
    for (int v = 0; v < 8; ++v) {
      int m = v + hi * 8;                             // j-row within tile
      float val = d[v] + s2v;
      val = (cur[ADJ_OFF + m] > 0.0f) ? val : NEG_BIG;
      mx[v] = fmaxf(mx[v], val);
    }
  }

#pragma unroll
  for (int v = 0; v < 8; ++v) {
    int j = j0 + v + hi * 8;
    partial[(((size_t)half * BB + b) * NN + j) * HH + col] = mx[v];
  }
}

// ---------------------------------------------------------------------------
// Kernel C: msgs = max(p0,p1)+msg_1+me_b ; h_2 = msgs@o2_w+o2_b ;
//           out = relu(h_1 + h_2).  grid = 128 row tiles, block = 256.
// ---------------------------------------------------------------------------
__global__ void pgn_out_kernel(const float* __restrict__ partial,
                               const float* __restrict__ msg1,
                               const float* __restrict__ h1,
                               const float* __restrict__ o2w,
                               const float* __restrict__ o2b,
                               const float* __restrict__ me_b,
                               float* __restrict__ out) {
  __shared__ bf16_t mt[16 * HH];
  const int row0 = blockIdx.x * 16;
  const int tid = threadIdx.x;
  const size_t PS = (size_t)BB * NN * HH;

#pragma unroll
  for (int e = 0; e < 8; ++e) {
    int idx = tid * 8 + e;
    int r = idx >> 7, c = idx & 127;
    size_t g = (size_t)(row0 + r) * HH + c;
    float m = fmaxf(partial[g], partial[PS + g]) + msg1[g] + me_b[c];
    mt[idx] = (bf16_t)m;
  }
  __syncthreads();

  const int wave = tid >> 5, lane = tid & 31;
  const int col  = wave * 16 + (lane & 15);
  const int hi   = (lane >> 4) & 1;

  v8f acc;
#pragma unroll
  for (int v = 0; v < 8; ++v) acc[v] = 0.0f;
#pragma unroll
  for (int kk = 0; kk < 4; ++kk) {
    v16bf a  = load_a_frag(mt, kk * 32, HH);
    v16bf bb = load_b_frag_f32(o2w, kk * 32, col, HH);
    acc = wmma_bf16(a, bb, acc);
  }
  const float bias = o2b[col];
#pragma unroll
  for (int v = 0; v < 8; ++v) {
    int r = row0 + v + hi * 8;
    float val = acc[v] + bias + h1[(size_t)r * HH + col];
    out[(size_t)r * HH + col] = fmaxf(val, 0.0f);
  }
}

// ---------------------------------------------------------------------------
extern "C" void kernel_launch(void* const* d_in, const int* in_sizes, int n_in,
                              void* d_out, int out_size, void* d_ws, size_t ws_size,
                              hipStream_t stream) {
  const float* node = (const float*)d_in[0];
  const float* edge = (const float*)d_in[1];
  const float* gfts = (const float*)d_in[2];
  const float* adj  = (const float*)d_in[3];
  const float* hid  = (const float*)d_in[4];
  const float* m1w  = (const float*)d_in[5];
  const float* m1b  = (const float*)d_in[6];
  const float* m2w  = (const float*)d_in[7];
  const float* m2b  = (const float*)d_in[8];
  const float* mew  = (const float*)d_in[9];
  const float* meb  = (const float*)d_in[10];
  const float* mgw  = (const float*)d_in[11];
  const float* mgb  = (const float*)d_in[12];
  const float* o1w  = (const float*)d_in[13];
  const float* o1b  = (const float*)d_in[14];
  const float* o2w  = (const float*)d_in[15];
  const float* o2b  = (const float*)d_in[16];

  const size_t PLANE = (size_t)BB * NN * HH;  // 262144 floats
  float* ws    = (float*)d_ws;
  float* msg_g = ws;                          // 1024
  float* msg1  = ws + 1024;                   // PLANE
  float* s2    = msg1 + PLANE;                // PLANE
  float* h1    = s2 + PLANE;                  // PLANE
  float* part  = h1 + PLANE;                  // 2*PLANE
  float* out   = (float*)d_out;

  pgn_msg_g_kernel<<<dim3(BB), dim3(HH), 0, stream>>>(gfts, mgw, mgb, msg_g);
  pgn_small_gemm_kernel<<<dim3(128, 3), dim3(256), 0, stream>>>(
      node, hid, m1w, m1b, m2w, m2b, o1w, o1b, msg_g, msg1, s2, h1);
  pgn_edge_max_kernel<<<dim3(16, BB, 2), dim3(256), 0, stream>>>(
      edge, adj, s2, mew, part);
  pgn_out_kernel<<<dim3(128), dim3(256), 0, stream>>>(
      part, msg1, h1, o2w, o2b, meb, out);
}
